// StyledConv_41394894799078
// MI455X (gfx1250) — matrix operand order repaired and checked by reading, not compiled, not run on hardware
//
#include <hip/hip_runtime.h>
#include <cmath>

typedef __attribute__((ext_vector_type(16))) __bf16 v16bf;
typedef __attribute__((ext_vector_type(8)))  __bf16 v8bf;
typedef __attribute__((ext_vector_type(8)))  float  v8f;
typedef __attribute__((ext_vector_type(4)))  float  f32x4;
typedef __attribute__((ext_vector_type(8)))  unsigned short u16x8;

#define NB    8
#define CIN   512
#define COUT  512
#define HW    64
#define HP    66     // padded spatial (1-pixel zero halo)
#define KTOT  4608   // 9 * 512, K order: tap*512 + cin
#define NCHUNK 144   // KTOT / 32
#define LDSP  40     // padded LDS row stride (elements): 80B -> 16B aligned, bank-rotating

__device__ __forceinline__ unsigned short f32_to_bf16(float f) {
  union { float f; unsigned u; } a; a.f = f;
  unsigned r = a.u + 0x7FFFu + ((a.u >> 16) & 1u);
  return (unsigned short)(r >> 16);
}

// LDS byte offset of a __shared__ pointer (addrspace(3) value == offset in LDS)
__device__ __forceinline__ unsigned lds_addr_of(const unsigned short* p) {
  typedef __attribute__((address_space(3))) const unsigned short* lds_ptr;
  return (unsigned)(unsigned long long)(lds_ptr)p;
}

// Async DMA: global -> LDS, 16 bytes per lane, tracked by ASYNCcnt
__device__ __forceinline__ void async_load_b128(unsigned lds_off, const unsigned short* g) {
  asm volatile("global_load_async_to_lds_b128 %0, %1, off"
               :: "v"(lds_off), "v"((unsigned long long)(size_t)g)
               : "memory");
}
__device__ __forceinline__ void wait_async0() {
  asm volatile("s_wait_asynccnt 0" ::: "memory");
}

// A fragment (16x32 bf16): lane(hi) holds K = hi*16 .. hi*16+15 (contiguous)
__device__ __forceinline__ v16bf frag_a(const unsigned short* p) {
  v8bf lo = *(const v8bf*)p;
  v8bf hi = *(const v8bf*)(p + 8);
  return __builtin_shufflevector(lo, hi, 0,1,2,3,4,5,6,7,8,9,10,11,12,13,14,15);
}
// B fragment (32x16 bf16): lane(hi) holds K = hi*16 .. hi*16+15 (contiguous)
__device__ __forceinline__ v16bf frag_b(const unsigned short* p) {
  v8bf lo = *(const v8bf*)p;
  v8bf hi = *(const v8bf*)(p + 8);
  return __builtin_shufflevector(lo, hi, 0,1,2,3,4,5,6,7,8,9,10,11,12,13,14,15);
}

// ---------------- zero padded activation buffers (halo must be 0 every call)
__global__ void zero_u16_kernel(unsigned short* __restrict__ p, int n8) {
  int i = blockIdx.x * blockDim.x + threadIdx.x;
  if (i >= n8) return;
  const u16x8 z = {0,0,0,0,0,0,0,0};
  *(u16x8*)(p + (size_t)i * 8) = z;
}

// ---------------- NCHW f32 -> padded NHWC bf16 (style)
__global__ void cvt_nhwc_kernel(const float* __restrict__ in,
                                unsigned short* __restrict__ out) {
  int idx = blockIdx.x * blockDim.x + threadIdx.x;   // NB*CIN*HW*(HW/4)
  int w4 = idx & 15;
  int h  = (idx >> 4) & 63;
  int c  = (idx >> 10) & 511;
  int b  = idx >> 19;
  f32x4 v = *(const f32x4*)(in + (((size_t)b * CIN + c) * 64 + h) * 64 + w4 * 4);
  size_t obase = (((size_t)b * HP + h + 1) * HP + w4 * 4 + 1) * CIN + c;
  out[obase]            = f32_to_bf16(v[0]);
  out[obase + CIN]      = f32_to_bf16(v[1]);
  out[obase + 2 * CIN]  = f32_to_bf16(v[2]);
  out[obase + 3 * CIN]  = f32_to_bf16(v[3]);
}

// ---------------- Weight transpose: [O][I][3][3] f32 -> [O][tap*512+cin] bf16 (pre-scaled)
__global__ void prep_weights(const float* __restrict__ w,
                             const float* __restrict__ gw,
                             const float* __restrict__ bw,
                             unsigned short* __restrict__ wT,
                             unsigned short* __restrict__ gT,
                             unsigned short* __restrict__ bT,
                             float scale) {
  int idx = blockIdx.x * blockDim.x + threadIdx.x;
  const int total = COUT * KTOT;
  if (idx >= total) return;
  int n   = idx / KTOT;
  int rem = idx - n * KTOT;
  int r   = rem / CIN;
  int c   = rem - r * CIN;
  int src = (n * CIN + c) * 9 + r;
  int dst = n * KTOT + r * CIN + c;
  wT[dst] = f32_to_bf16(w[src]  * scale);
  gT[dst] = f32_to_bf16(gw[src] * scale);
  bT[dst] = f32_to_bf16(bw[src] * scale);
}

// ---------------- InstanceNorm stats: one block per (b,c)
__global__ void instnorm_stats(const float* __restrict__ x,
                               float* __restrict__ mean,
                               float* __restrict__ rstd) {
  __shared__ float ss[256], sq[256];
  const int bc = blockIdx.x;
  const float* p = x + (size_t)bc * (HW * HW);
  float s = 0.f, q = 0.f;
  for (int i = threadIdx.x * 4; i < HW * HW; i += 1024) {
    f32x4 v = *(const f32x4*)(p + i);
#pragma unroll
    for (int e = 0; e < 4; ++e) { s += v[e]; q += v[e] * v[e]; }
  }
  ss[threadIdx.x] = s; sq[threadIdx.x] = q;
  __syncthreads();
  for (int off = 128; off > 0; off >>= 1) {
    if (threadIdx.x < off) {
      ss[threadIdx.x] += ss[threadIdx.x + off];
      sq[threadIdx.x] += sq[threadIdx.x + off];
    }
    __syncthreads();
  }
  if (threadIdx.x == 0) {
    float m = ss[0] * (1.f / 4096.f);
    float v = sq[0] * (1.f / 4096.f) - m * m;
    mean[bc] = m;
    rstd[bc] = rsqrtf(v + 1e-5f);
  }
}

// ---------------- Kernel B: gamma/beta conv GEMM (dual-B), fused IN + modulation -> xm
__global__ __launch_bounds__(256)
void style_mod_kernel(const float* __restrict__ x,
                      const unsigned short* __restrict__ styleb,   // padded NHWC bf16
                      const unsigned short* __restrict__ gT,
                      const unsigned short* __restrict__ bT,
                      const float* __restrict__ gamma_b,
                      const float* __restrict__ beta_b,
                      const float* __restrict__ mean,
                      const float* __restrict__ rstd,
                      unsigned short* __restrict__ xm) {            // padded NHWC bf16
  __shared__ unsigned short sA[2][64 * LDSP];
  __shared__ unsigned short sBg[2][128 * LDSP];
  __shared__ unsigned short sBb[2][128 * LDSP];

  const int tid = threadIdx.x;
  const int b   = blockIdx.x >> 6;
  const int h   = blockIdx.x & 63;
  const int n0  = blockIdx.y * 128;

  const int lane = tid & 31, wid = tid >> 5;
  const int wm = wid & 1, wn = wid >> 1;     // 2 x 4 wave grid
  const int l16 = lane & 15, hi = lane >> 4;

  const int akb = tid & 3;    // contiguous cin range akb*8..+7
  const int am  = tid >> 2;   // spatial (w) position for A fill
  const int bn  = tid >> 1;   // 0..127 channel row for B fill
  const int bpp = tid & 1;

  const v8f vz = {0.f,0.f,0.f,0.f,0.f,0.f,0.f,0.f};
  v8f accg[2][2] = {{vz,vz},{vz,vz}}, accb[2][2] = {{vz,vz},{vz,vz}};

  auto fill_async = [&](int buf, int kk) {
    const int r  = kk >> 4;
    const int c0 = (kk & 15) << 5;
    // padded coords: hp = h + r/3 in [0,65], wp = am + r%3 in [0,65]
    const size_t ga = (((size_t)b * HP + (h + r / 3)) * HP + (am + r % 3)) * CIN
                      + c0 + akb * 8;
    async_load_b128(lds_addr_of(&sA[buf][am * LDSP + akb * 8]), styleb + ga);
    const size_t g0 = (size_t)(n0 + bn) * KTOT + r * 512 + c0 + bpp * 8;
    const unsigned lg = lds_addr_of(&sBg[buf][bn * LDSP + bpp * 8]);
    const unsigned lb = lds_addr_of(&sBb[buf][bn * LDSP + bpp * 8]);
    async_load_b128(lg,      gT + g0);
    async_load_b128(lg + 32, gT + g0 + 16);
    async_load_b128(lb,      bT + g0);
    async_load_b128(lb + 32, bT + g0 + 16);
  };

  fill_async(0, 0);
  wait_async0();
  __syncthreads();

  for (int k = 0; k < NCHUNK; ++k) {
    const int cur = k & 1;
    const bool more = (k + 1) < NCHUNK;
    if (more) fill_async(cur ^ 1, k + 1);

    v16bf af[2], bg[2], bb2[2];
#pragma unroll
    for (int i = 0; i < 2; ++i)
      af[i] = frag_a(&sA[cur][(wm * 32 + i * 16 + l16) * LDSP + hi * 16]);
#pragma unroll
    for (int j = 0; j < 2; ++j) {
      bg[j]  = frag_b(&sBg[cur][(wn * 32 + j * 16 + l16) * LDSP + hi * 16]);
      bb2[j] = frag_b(&sBb[cur][(wn * 32 + j * 16 + l16) * LDSP + hi * 16]);
    }
#pragma unroll
    for (int i = 0; i < 2; ++i)
#pragma unroll
      for (int j = 0; j < 2; ++j) {
        accg[i][j] = __builtin_amdgcn_wmma_f32_16x16x32_bf16(
            false, af[i], false, bg[j],  (short)0, accg[i][j], false, false);
        accb[i][j] = __builtin_amdgcn_wmma_f32_16x16x32_bf16(
            false, af[i], false, bb2[j], (short)0, accb[i][j], false, false);
      }
    wait_async0();
    __syncthreads();
  }

  // Epilogue: xm(padded NHWC) = ((x - mu) * rstd) * (gamma + gb) + (beta + bb)
  const size_t spat = ((size_t)b * HP + h + 1) * HP + 1;   // padded (b,h) base
#pragma unroll
  for (int j = 0; j < 2; ++j) {
    const int c = n0 + wn * 32 + j * 16 + l16;
    const float mu = mean[b * CIN + c];
    const float rs = rstd[b * CIN + c];
    const float gb = gamma_b[c];
    const float bb = beta_b[c];
    const size_t rowoff = (((size_t)b * CIN + c) * 64 + h) * 64;  // x is NCHW
#pragma unroll
    for (int i = 0; i < 2; ++i) {
      const v8f ag = accg[i][j], ab = accb[i][j];
      const int m0 = wm * 32 + i * 16 + hi * 8;
      f32x4 x0 = *(const f32x4*)(x + rowoff + m0);
      f32x4 x1 = *(const f32x4*)(x + rowoff + m0 + 4);
#pragma unroll
      for (int v = 0; v < 4; ++v) {
        float xn = (x0[v] - mu) * rs;
        xm[(spat + m0 + v) * CIN + c] = f32_to_bf16(xn * (ag[v] + gb) + (ab[v] + bb));
      }
#pragma unroll
      for (int v = 4; v < 8; ++v) {
        float xn = (x1[v - 4] - mu) * rs;
        xm[(spat + m0 + v) * CIN + c] = f32_to_bf16(xn * (ag[v] + gb) + (ab[v] + bb));
      }
    }
  }
}

// ---------------- Kernel C: main modulated conv GEMM, fused bias + LeakyReLU*sqrt(2)
__global__ __launch_bounds__(256)
void main_conv_kernel(const unsigned short* __restrict__ xmb,   // padded NHWC bf16
                      const unsigned short* __restrict__ wT,
                      const float* __restrict__ act_bias,
                      float* __restrict__ out) {                 // NCHW f32
  __shared__ unsigned short sA[2][64 * LDSP];
  __shared__ unsigned short sB[2][128 * LDSP];

  const int tid = threadIdx.x;
  const int b   = blockIdx.x >> 6;
  const int h   = blockIdx.x & 63;
  const int n0  = blockIdx.y * 128;

  const int lane = tid & 31, wid = tid >> 5;
  const int wm = wid & 1, wn = wid >> 1;
  const int l16 = lane & 15, hi = lane >> 4;

  const int akb = tid & 3;
  const int am  = tid >> 2;
  const int bn  = tid >> 1;
  const int bpp = tid & 1;

  const v8f vz = {0.f,0.f,0.f,0.f,0.f,0.f,0.f,0.f};
  v8f acc[2][2] = {{vz,vz},{vz,vz}};

  auto fill_async = [&](int buf, int kk) {
    const int r  = kk >> 4;
    const int c0 = (kk & 15) << 5;
    const size_t ga = (((size_t)b * HP + (h + r / 3)) * HP + (am + r % 3)) * CIN
                      + c0 + akb * 8;
    async_load_b128(lds_addr_of(&sA[buf][am * LDSP + akb * 8]), xmb + ga);
    const size_t g0 = (size_t)(n0 + bn) * KTOT + r * 512 + c0 + bpp * 8;
    const unsigned lb = lds_addr_of(&sB[buf][bn * LDSP + bpp * 8]);
    async_load_b128(lb,      wT + g0);
    async_load_b128(lb + 32, wT + g0 + 16);
  };

  fill_async(0, 0);
  wait_async0();
  __syncthreads();

  for (int k = 0; k < NCHUNK; ++k) {
    const int cur = k & 1;
    const bool more = (k + 1) < NCHUNK;
    if (more) fill_async(cur ^ 1, k + 1);

    v16bf af[2], bf[2];
#pragma unroll
    for (int i = 0; i < 2; ++i)
      af[i] = frag_a(&sA[cur][(wm * 32 + i * 16 + l16) * LDSP + hi * 16]);
#pragma unroll
    for (int j = 0; j < 2; ++j)
      bf[j] = frag_b(&sB[cur][(wn * 32 + j * 16 + l16) * LDSP + hi * 16]);
#pragma unroll
    for (int i = 0; i < 2; ++i)
#pragma unroll
      for (int j = 0; j < 2; ++j)
        acc[i][j] = __builtin_amdgcn_wmma_f32_16x16x32_bf16(
            false, af[i], false, bf[j], (short)0, acc[i][j], false, false);

    wait_async0();
    __syncthreads();
  }

#pragma unroll
  for (int j = 0; j < 2; ++j) {
    const int c = n0 + wn * 32 + j * 16 + l16;
    const float bias = act_bias[c];
    const size_t rowoff = (((size_t)b * COUT + c) * 64 + h) * 64;
#pragma unroll
    for (int i = 0; i < 2; ++i) {
      const v8f a = acc[i][j];
      const int m0 = wm * 32 + i * 16 + hi * 8;
      f32x4 o0, o1;
#pragma unroll
      for (int v = 0; v < 4; ++v) {
        float t = a[v] + bias;
        o0[v] = (t >= 0.f ? t : 0.2f * t) * 1.4142135623730951f;
      }
#pragma unroll
      for (int v = 4; v < 8; ++v) {
        float t = a[v] + bias;
        o1[v - 4] = (t >= 0.f ? t : 0.2f * t) * 1.4142135623730951f;
      }
      *(f32x4*)(out + rowoff + m0)     = o0;
      *(f32x4*)(out + rowoff + m0 + 4) = o1;
    }
  }
}

extern "C" void kernel_launch(void* const* d_in, const int* in_sizes, int n_in,
                              void* d_out, int out_size, void* d_ws, size_t ws_size,
                              hipStream_t stream) {
  (void)in_sizes; (void)n_in; (void)out_size; (void)ws_size;
  const float* x        = (const float*)d_in[0];
  const float* style    = (const float*)d_in[1];
  const float* w        = (const float*)d_in[2];
  const float* gamma_w  = (const float*)d_in[3];
  const float* gamma_b  = (const float*)d_in[4];
  const float* beta_w   = (const float*)d_in[5];
  const float* beta_b   = (const float*)d_in[6];
  const float* act_bias = (const float*)d_in[7];
  float* out = (float*)d_out;

  char* ws = (char*)d_ws;
  unsigned short* wT     = (unsigned short*)(ws + 0);          // 4718592 B
  unsigned short* gT     = (unsigned short*)(ws + 4718592);
  unsigned short* bT     = (unsigned short*)(ws + 9437184);
  float*          mean   = (float*)(ws + 14155776);            // 16 KB
  float*          rstd   = (float*)(ws + 14172160);            // 16 KB
  unsigned short* xm     = (unsigned short*)(ws + 14188544);   // 8*66*66*512*2 = 35684352 B
  unsigned short* styleb = (unsigned short*)(ws + 49872896);   // 35684352 B (adjacent)

  const float scale = 1.0f / sqrtf(4608.0f);  // eq_scale == w_scale

  // zero both padded buffers (xm + styleb are contiguous)
  const int padN8 = 2 * NB * HP * HP * CIN / 8;   // 4,460,544 x 16B
  zero_u16_kernel<<<(padN8 + 255) / 256, 256, 0, stream>>>(xm, padN8);

  const int nThr = NB * CIN * HW * (HW / 4);  // 4,194,304
  cvt_nhwc_kernel<<<nThr / 256, 256, 0, stream>>>(style, styleb);
  const int totalW = COUT * KTOT;
  prep_weights<<<(totalW + 255) / 256, 256, 0, stream>>>(w, gamma_w, beta_w, wT, gT, bT, scale);
  instnorm_stats<<<NB * CIN, 256, 0, stream>>>(x, mean, rstd);
  style_mod_kernel<<<dim3(NB * HW, CIN / 128), 256, 0, stream>>>(
      x, styleb, gT, bT, gamma_b, beta_b, mean, rstd, xm);
  main_conv_kernel<<<dim3(NB * HW, COUT / 128), 256, 0, stream>>>(xm, wT, act_bias, out);
}
